// MoE_73658689126739
// MI455X (gfx1250) — matrix operand order repaired and checked by reading, not compiled, not run on hardware
//
#include <hip/hip_runtime.h>
#include <hip/hip_bf16.h>
#include <math.h>

#define N_TOK 4096
#define DIM   1024
#define HID   2048
#define NEXP  8

typedef __attribute__((ext_vector_type(16))) _Float16 v16h;
typedef __attribute__((ext_vector_type(8)))  _Float16 v8h;
typedef __attribute__((ext_vector_type(4)))  _Float16 v4h;
typedef __attribute__((ext_vector_type(8)))  float    v8f;
typedef __attribute__((ext_vector_type(4)))  float    vf4;
typedef __attribute__((ext_vector_type(4)))  int      v4i;

#define LDA 40  // padded LDS row stride in f16 elems (80B = multiple of 16B)

#if __has_builtin(__builtin_amdgcn_global_load_async_to_lds_b128)
#define HAVE_ASYNC_LDS 1
// per-lane async copy of 16B global -> LDS, tracked by ASYNCcnt
#define ASYNC_B128(gsrc, ldst)                                                \
  __builtin_amdgcn_global_load_async_to_lds_b128(                             \
      (__attribute__((address_space(1))) v4i*)(uintptr_t)(gsrc),              \
      (__attribute__((address_space(3))) v4i*)(unsigned)(uintptr_t)(ldst),    \
      0, 0)
#else
#define HAVE_ASYNC_LDS 0
#endif

// ---- WMMA fragment loaders (ISA 7.12.2 layouts) -------------------------
__device__ __forceinline__ v16h frag_a(const _Float16* lds, int rowBase, int lane) {
  int r  = rowBase + (lane & 15);
  int kh = (lane >> 4) << 3;                       // 0 or 8
  v8h g1 = *(const v8h*)(lds + r * LDA + kh);      // K = kh .. kh+7
  v8h g2 = *(const v8h*)(lds + r * LDA + 16 + kh); // K = 16+kh .. 23+kh
  return __builtin_shufflevector(g1, g2, 0,1,2,3,4,5,6,7,8,9,10,11,12,13,14,15);
}
__device__ __forceinline__ v16h frag_b(const _Float16* lds, int colBase, int lane) {
  int c  = colBase + (lane & 15);
  int kb = (lane >> 4) << 3;                       // 0 or 8
  v8h g1 = *(const v8h*)(lds + c * LDA + kb);
  v8h g2 = *(const v8h*)(lds + c * LDA + 16 + kb);
  return __builtin_shufflevector(g1, g2, 0,1,2,3,4,5,6,7,8,9,10,11,12,13,14,15);
}

__device__ __forceinline__ uint32_t pk2(float lo, float hi) {
  union { _Float16 h[2]; uint32_t u; } t;
  t.h[0] = (_Float16)lo; t.h[1] = (_Float16)hi;
  return t.u;
}

__device__ __forceinline__ float gelu_tanh(float x) {
  const float c0 = 0.7978845608028654f;   // sqrt(2/pi)
  const float c1 = 0.044715f;
  return 0.5f * x * (1.0f + tanhf(c0 * (x + c1 * x * x * x)));
}

// 8x WMMA chain on preloaded fragments (dense issue burst for the XDL pipe)
#define WMMA8(A0, A1, B0, B1, B2, B3, ACC)                                                            \
  do {                                                                                                \
    ACC[0][0] = __builtin_amdgcn_wmma_f32_16x16x32_f16(false, A0, false, B0, (short)0, ACC[0][0], false, false); \
    ACC[1][0] = __builtin_amdgcn_wmma_f32_16x16x32_f16(false, A1, false, B0, (short)0, ACC[1][0], false, false); \
    ACC[0][1] = __builtin_amdgcn_wmma_f32_16x16x32_f16(false, A0, false, B1, (short)0, ACC[0][1], false, false); \
    ACC[1][1] = __builtin_amdgcn_wmma_f32_16x16x32_f16(false, A1, false, B1, (short)0, ACC[1][1], false, false); \
    ACC[0][2] = __builtin_amdgcn_wmma_f32_16x16x32_f16(false, A0, false, B2, (short)0, ACC[0][2], false, false); \
    ACC[1][2] = __builtin_amdgcn_wmma_f32_16x16x32_f16(false, A1, false, B2, (short)0, ACC[1][2], false, false); \
    ACC[0][3] = __builtin_amdgcn_wmma_f32_16x16x32_f16(false, A0, false, B3, (short)0, ACC[0][3], false, false); \
    ACC[1][3] = __builtin_amdgcn_wmma_f32_16x16x32_f16(false, A1, false, B3, (short)0, ACC[1][3], false, false); \
  } while (0)

// ---- kernel 0: zero per-expert counters ---------------------------------
__global__ void moe_init(int* __restrict__ counts) {
  if (threadIdx.x < NEXP) counts[threadIdx.x] = 0;
}

// ---- kernel 1: gating (logits, top-2 softmax, expert token lists) -------
__global__ __launch_bounds__(256)
void moe_gate(const float* __restrict__ x, const float* __restrict__ wg,
              int* __restrict__ counts, int* __restrict__ lists,
              float* __restrict__ gates) {
  int n    = blockIdx.x * 8 + (threadIdx.x >> 5);
  int lane = threadIdx.x & 31;
  if (n >= N_TOK) return;
  float a[NEXP] = {0,0,0,0,0,0,0,0};
  const float* xr = x + (size_t)n * DIM;
  for (int d = lane; d < DIM; d += 32) {
    float xv = xr[d];
    const vf4 wlo = *(const vf4*)(wg + d * NEXP);
    const vf4 whi = *(const vf4*)(wg + d * NEXP + 4);
    a[0] += xv * wlo[0]; a[1] += xv * wlo[1]; a[2] += xv * wlo[2]; a[3] += xv * wlo[3];
    a[4] += xv * whi[0]; a[5] += xv * whi[1]; a[6] += xv * whi[2]; a[7] += xv * whi[3];
  }
#pragma unroll
  for (int e = 0; e < NEXP; ++e)
#pragma unroll
    for (int off = 16; off > 0; off >>= 1)
      a[e] += __shfl_xor(a[e], off, 32);
  if (lane == 0) {
    int e0 = 0; float v0 = a[0];
#pragma unroll
    for (int e = 1; e < NEXP; ++e) if (a[e] > v0) { v0 = a[e]; e0 = e; }
    int e1 = (e0 == 0) ? 1 : 0; float v1 = a[e1];
#pragma unroll
    for (int e = 0; e < NEXP; ++e) if (e != e0 && a[e] > v1) { v1 = a[e]; e1 = e; }
    float g0 = 1.0f / (1.0f + __expf(v1 - v0));   // softmax over {v0,v1}
    gates[2 * n + 0] = g0;
    gates[2 * n + 1] = 1.0f - g0;
    int p0 = atomicAdd(&counts[e0], 1); lists[e0 * N_TOK + p0] = n * 2 + 0;
    int p1 = atomicAdd(&counts[e1], 1); lists[e1 * N_TOK + p1] = n * 2 + 1;
  }
}

// ---- kernel 2: expert FFN layer 1: h = gelu(x @ w1[e] + b1[e]) ----------
// block tile 128 tokens x 128 hidden; 8 waves (4 along M, 2 along N), each 32x64
__global__ __launch_bounds__(256)
void moe_ffn1(const float* __restrict__ x, const float* __restrict__ w1,
              const float* __restrict__ b1, const int* __restrict__ counts,
              const int* __restrict__ lists, _Float16* __restrict__ hbuf) {
  __shared__ _Float16 ldsA[128 * LDA];
  __shared__ _Float16 ldsB[128 * LDA];
  __shared__ int ldsEnt[128];

  const int e   = blockIdx.z;
  const int cnt = counts[e];
  const int m0  = blockIdx.x * 128;
  if (m0 >= cnt) return;
  const int h0  = blockIdx.y * 128;

  const int tid  = threadIdx.x;
  const int lane = tid & 31;
  const int wv   = tid >> 5;
  const int mw   = (wv & 3) * 32;   // wave row base
  const int nw   = (wv >> 2) * 64;  // wave col base

  if (tid < 128) {
    int m = m0 + tid;
    ldsEnt[tid] = (m < cnt) ? lists[e * N_TOK + m] : -1;
  }
  __syncthreads();

  // A: 128 rows x 32 k; thread covers 4 rows (rg + 32p), 4 k at cg
  const int rg = tid >> 3;
  const int cg = (tid & 7) * 4;
  const float* aPtr[4];
#pragma unroll
  for (int p = 0; p < 4; ++p) {
    int ent = ldsEnt[rg + p * 32];
    int tok = (ent >= 0) ? (ent >> 1) : 0;
    aPtr[p] = x + (size_t)tok * DIM + cg;
  }
  // B: 32 k x 128 cols; thread covers rows {kA,kA+1,kA+16,kA+17} x cols cb..cb+3
  const int cb = (tid & 31) * 4;
  const int kA = (tid >> 5) * 2;
  const float* w1e  = w1 + (size_t)e * DIM * HID;
  const float* bPtr = w1e + h0 + cb;

  vf4 ar[4], br[4];
  auto ldt = [&](int k0) {
#pragma unroll
    for (int p = 0; p < 4; ++p) ar[p] = *(const vf4*)(aPtr[p] + k0);
    br[0] = *(const vf4*)(bPtr + (size_t)(k0 + kA + 0)  * HID);
    br[1] = *(const vf4*)(bPtr + (size_t)(k0 + kA + 1)  * HID);
    br[2] = *(const vf4*)(bPtr + (size_t)(k0 + kA + 16) * HID);
    br[3] = *(const vf4*)(bPtr + (size_t)(k0 + kA + 17) * HID);
  };

  v8f acc[2][4] = {{{},{},{},{}}, {{},{},{},{}}};

  ldt(0);
  for (int k0 = 0; k0 < DIM; k0 += 32) {
    __syncthreads();
#pragma unroll
    for (int p = 0; p < 4; ++p) {
      v4h hv; hv[0] = (_Float16)ar[p][0]; hv[1] = (_Float16)ar[p][1];
              hv[2] = (_Float16)ar[p][2]; hv[3] = (_Float16)ar[p][3];
      *(v4h*)(ldsA + (rg + p * 32) * LDA + cg) = hv;
    }
#pragma unroll
    for (int i = 0; i < 4; ++i) {
      *(uint32_t*)(ldsB + (cb + i) * LDA + kA)      = pk2(br[0][i], br[1][i]);
      *(uint32_t*)(ldsB + (cb + i) * LDA + kA + 16) = pk2(br[2][i], br[3][i]);
    }
    __syncthreads();
    if (k0 + 32 < DIM) ldt(k0 + 32);   // overlap next tile loads with WMMA

    v16h a0  = frag_a(ldsA, mw + 0,  lane);
    v16h a1  = frag_a(ldsA, mw + 16, lane);
    v16h bf0 = frag_b(ldsB, nw + 0,  lane);
    v16h bf1 = frag_b(ldsB, nw + 16, lane);
    v16h bf2 = frag_b(ldsB, nw + 32, lane);
    v16h bf3 = frag_b(ldsB, nw + 48, lane);
    WMMA8(a0, a1, bf0, bf1, bf2, bf3, acc);
  }

  // epilogue: +bias, gelu, f16 store to hbuf[token*2+slot][h]
  const int hi = (lane >> 4) << 3;
  const int cl = lane & 15;
#pragma unroll
  for (int j = 0; j < 4; ++j) {
    int col = h0 + nw + j * 16 + cl;
    float bias = b1[e * HID + col];
#pragma unroll
    for (int i = 0; i < 2; ++i)
#pragma unroll
      for (int v = 0; v < 8; ++v) {
        int ml  = mw + i * 16 + hi + v;
        int ent = ldsEnt[ml];
        if (ent >= 0)
          hbuf[(size_t)ent * HID + col] = (_Float16)gelu_tanh(acc[i][j][v] + bias);
      }
  }
}

// ---- kernel 3: expert FFN layer 2: y = h @ w2[e] + b2[e] ----------------
// A tile (f16 activations) staged by CDNA5 async global->LDS when available,
// double-buffered; B tile register-staged (needs f32->f16 convert).
__global__ __launch_bounds__(256)
void moe_ffn2(const _Float16* __restrict__ hbuf, const float* __restrict__ w2,
              const float* __restrict__ b2, const int* __restrict__ counts,
              const int* __restrict__ lists, float* __restrict__ ybuf) {
  __shared__ _Float16 ldsA[2][128 * LDA];
  __shared__ _Float16 ldsB[128 * LDA];
  __shared__ int ldsEnt[128];

  const int e   = blockIdx.z;
  const int cnt = counts[e];
  const int m0  = blockIdx.x * 128;
  if (m0 >= cnt) return;
  const int d0  = blockIdx.y * 128;

  const int tid  = threadIdx.x;
  const int lane = tid & 31;
  const int wv   = tid >> 5;
  const int mw   = (wv & 3) * 32;
  const int nw   = (wv >> 2) * 64;

  if (tid < 128) {
    int m = m0 + tid;
    ldsEnt[tid] = (m < cnt) ? lists[e * N_TOK + m] : -1;
  }
  __syncthreads();

  // A: 128 rows x 32 k f16; thread covers 2 rows (rg + 64p), 8 k at cg
  const int rg = tid >> 2;
  const int cg = (tid & 3) * 8;
  const _Float16* hPtr[2];
#pragma unroll
  for (int p = 0; p < 2; ++p) {
    int ent = ldsEnt[rg + p * 64];
    int r   = (ent >= 0) ? ent : 0;
    hPtr[p] = hbuf + (size_t)r * HID + cg;
  }
  const int cb = (tid & 31) * 4;
  const int kA = (tid >> 5) * 2;
  const float* w2e  = w2 + (size_t)e * HID * DIM;
  const float* bPtr = w2e + d0 + cb;

  vf4 br[4];
  auto ldtB = [&](int k0) {
    br[0] = *(const vf4*)(bPtr + (size_t)(k0 + kA + 0)  * DIM);
    br[1] = *(const vf4*)(bPtr + (size_t)(k0 + kA + 1)  * DIM);
    br[2] = *(const vf4*)(bPtr + (size_t)(k0 + kA + 16) * DIM);
    br[3] = *(const vf4*)(bPtr + (size_t)(k0 + kA + 17) * DIM);
  };

  v8f acc[2][4] = {{{},{},{},{}}, {{},{},{},{}}};

#if HAVE_ASYNC_LDS
  auto acpA = [&](int k0, int buf) {      // async A tile: 2x b128 per thread
#pragma unroll
    for (int p = 0; p < 2; ++p)
      ASYNC_B128(hPtr[p] + k0, &ldsA[buf][(rg + p * 64) * LDA + cg]);
  };
  const int nT = HID / 32;
  acpA(0, 0);
  ldtB(0);
  for (int t = 0; t < nT; ++t) {
    const int k0 = t * 32;
    __syncthreads();                       // ldsB (and A buf parity) free
#pragma unroll
    for (int i = 0; i < 4; ++i) {
      *(uint32_t*)(ldsB + (cb + i) * LDA + kA)      = pk2(br[0][i], br[1][i]);
      *(uint32_t*)(ldsB + (cb + i) * LDA + kA + 16) = pk2(br[2][i], br[3][i]);
    }
    if (t + 1 < nT) {
      acpA(k0 + 32, (t + 1) & 1);          // async fill of next A buffer
      ldtB(k0 + 32);
      // async loads complete in order: <=2 outstanding => tile t's are done
      asm volatile("s_wait_asynccnt 0x2" ::: "memory");
    } else {
      asm volatile("s_wait_asynccnt 0x0" ::: "memory");
    }
    __syncthreads();

    const _Float16* bufA = ldsA[t & 1];
    v16h a0  = frag_a(bufA, mw + 0,  lane);
    v16h a1  = frag_a(bufA, mw + 16, lane);
    v16h bf0 = frag_b(ldsB, nw + 0,  lane);
    v16h bf1 = frag_b(ldsB, nw + 16, lane);
    v16h bf2 = frag_b(ldsB, nw + 32, lane);
    v16h bf3 = frag_b(ldsB, nw + 48, lane);
    WMMA8(a0, a1, bf0, bf1, bf2, bf3, acc);
  }
#else
  v8h ah[2];
  auto ldtA = [&](int k0) {
#pragma unroll
    for (int p = 0; p < 2; ++p) ah[p] = *(const v8h*)(hPtr[p] + k0);
  };
  ldtA(0); ldtB(0);
  for (int k0 = 0; k0 < HID; k0 += 32) {
    __syncthreads();
#pragma unroll
    for (int p = 0; p < 2; ++p)
      *(v8h*)(ldsA[0] + (rg + p * 64) * LDA + cg) = ah[p];
#pragma unroll
    for (int i = 0; i < 4; ++i) {
      *(uint32_t*)(ldsB + (cb + i) * LDA + kA)      = pk2(br[0][i], br[1][i]);
      *(uint32_t*)(ldsB + (cb + i) * LDA + kA + 16) = pk2(br[2][i], br[3][i]);
    }
    __syncthreads();
    if (k0 + 32 < HID) { ldtA(k0 + 32); ldtB(k0 + 32); }

    v16h a0  = frag_a(ldsA[0], mw + 0,  lane);
    v16h a1  = frag_a(ldsA[0], mw + 16, lane);
    v16h bf0 = frag_b(ldsB, nw + 0,  lane);
    v16h bf1 = frag_b(ldsB, nw + 16, lane);
    v16h bf2 = frag_b(ldsB, nw + 32, lane);
    v16h bf3 = frag_b(ldsB, nw + 48, lane);
    WMMA8(a0, a1, bf0, bf1, bf2, bf3, acc);
  }
#endif

  const int hi = (lane >> 4) << 3;
  const int cl = lane & 15;
#pragma unroll
  for (int j = 0; j < 4; ++j) {
    int col = d0 + nw + j * 16 + cl;
    float bias = b2[e * DIM + col];
#pragma unroll
    for (int i = 0; i < 2; ++i)
#pragma unroll
      for (int v = 0; v < 8; ++v) {
        int ml  = mw + i * 16 + hi + v;
        int ent = ldsEnt[ml];
        if (ent >= 0)
          ybuf[(size_t)ent * DIM + col] = acc[i][j][v] + bias;
      }
  }
}

// ---- kernel 4: combine: out[n] = g0*y[2n] + g1*y[2n+1] ------------------
__global__ __launch_bounds__(256)
void moe_combine(const float* __restrict__ ybuf, const float* __restrict__ gates,
                 float* __restrict__ out) {
  int i = blockIdx.x * 256 + threadIdx.x;      // one float4 each
  if (i >= N_TOK * (DIM / 4)) return;
  int n = i >> 8;                              // DIM/4 == 256
  int c = (i & 255) * 4;
  float g0 = gates[2 * n], g1 = gates[2 * n + 1];
  const vf4 y0 = *(const vf4*)(ybuf + (size_t)(2 * n) * DIM + c);
  const vf4 y1 = *(const vf4*)(ybuf + (size_t)(2 * n + 1) * DIM + c);
  vf4 o = g0 * y0 + g1 * y1;
  *(vf4*)(out + (size_t)n * DIM + c) = o;
}

extern "C" void kernel_launch(void* const* d_in, const int* in_sizes, int n_in,
                              void* d_out, int out_size, void* d_ws, size_t ws_size,
                              hipStream_t stream) {
  const float* x  = (const float*)d_in[0];
  const float* wg = (const float*)d_in[1];
  const float* w1 = (const float*)d_in[2];
  const float* b1 = (const float*)d_in[3];
  const float* w2 = (const float*)d_in[4];
  const float* b2 = (const float*)d_in[5];
  float* out = (float*)d_out;

  char* ws = (char*)d_ws;
  int*      counts = (int*)ws;                                   // 8 ints
  int*      lists  = (int*)(ws + 256);                           // E*N ints (128KB)
  float*    gates  = (float*)(ws + 256 + NEXP * N_TOK * 4);      // N*2 f32 (32KB)
  _Float16* hbuf   = (_Float16*)(ws + (1u << 20));               // 8192 x 2048 f16 (32MB)
  float*    ybuf   = (float*)(ws + (1u << 20) +
                              (size_t)N_TOK * 2 * HID * 2);      // 8192 x 1024 f32 (32MB)

  moe_init<<<1, 32, 0, stream>>>(counts);
  moe_gate<<<N_TOK / 8, 256, 0, stream>>>(x, wg, counts, lists, gates);
  moe_ffn1<<<dim3(N_TOK / 128, HID / 128, NEXP), 256, 0, stream>>>(x, w1, b1, counts, lists, hbuf);
  moe_ffn2<<<dim3(N_TOK / 128, DIM / 128, NEXP), 256, 0, stream>>>(hbuf, w2, b2, counts, lists, ybuf);
  moe_combine<<<(N_TOK * (DIM / 4)) / 256, 256, 0, stream>>>(ybuf, gates, out);
}
